// Attn_9637906612873
// MI455X (gfx1250) — matrix operand order repaired and checked by reading, not compiled, not run on hardware
//
#include <hip/hip_runtime.h>
#include <hip/hip_bf16.h>

// ---------------------------------------------------------------------------
// Problem dims (fixed by the reference)
#define BQ   32
#define TQS  1024
#define LQ   2048
#define DQ   1024
#define DK   1024
#define DV   1024
#define DOUT 1024

typedef _Float16 half_t;
typedef __attribute__((ext_vector_type(16))) _Float16 v16h;
typedef __attribute__((ext_vector_type(8)))  _Float16 v8h;
typedef __attribute__((ext_vector_type(8)))  float    v8f;
typedef __attribute__((ext_vector_type(4)))  float    v4f;
typedef __attribute__((ext_vector_type(4)))  int      v4i;

// Tiling: block = 256 threads = 8 waves. Block tile 256x64, wave tile 32x64.
// B tile (64 out-cols x 32 k) staged in LDS, double buffered, rows padded to
// 80 B (16B-aligned, spreads banks for ds_load_b128 fragment reads).
#define LDS_STRIDE 40              // halfs per LDS B row (32 data + 8 pad)
#define BROWS      64              // B-tile rows (= output columns per block)
#define BUF_HALFS  (BROWS * LDS_STRIDE)

// ---------------------------------------------------------------------------
// CDNA5 async global->LDS copy (ASYNCcnt path), with safe fallback.
// Probed signature: (v4i addrspace(1)* src, v4i addrspace(3)* dst, Imm off, Imm cpol)
typedef __attribute__((address_space(1))) v4i gv4i_t;
typedef __attribute__((address_space(3))) v4i lv4i_t;

__device__ __forceinline__ void async_copy16(half_t* dst_lds, const half_t* src_g) {
#if __has_builtin(__builtin_amdgcn_global_load_async_to_lds_b128)
  __builtin_amdgcn_global_load_async_to_lds_b128(
      (gv4i_t*)src_g, (lv4i_t*)dst_lds, 0, 0);
#else
  *(v8h*)dst_lds = *(const v8h*)src_g;   // global_load + ds_store fallback
#endif
}

__device__ __forceinline__ void async_wait_all() {
#if __has_builtin(__builtin_amdgcn_global_load_async_to_lds_b128)
#if __has_builtin(__builtin_amdgcn_s_wait_asynccnt)
  __builtin_amdgcn_s_wait_asynccnt(0);
#else
  asm volatile("s_wait_asynccnt 0x0" ::: "memory");
#endif
#endif
}

// ---------------------------------------------------------------------------
// Fragment loaders matching CDNA5 WMMA VGPR layouts (CDNA5 ISA)
//
// A (16x32 f16, M x K): lane l holds row m = l&15.
//   halves 0..7  -> k = (l>=16 ? 8:0) + 0..7 ; halves 8..15 -> +16
__device__ __forceinline__ v16h load_frag_a(const half_t* __restrict__ base,
                                            int ld, int row0, int k0, int lane) {
  int m  = lane & 15;
  int hi = lane >> 4;
  const half_t* p = base + (size_t)(row0 + m) * ld + k0 + hi * 8;
  union { v16h v; v8h h[2]; } u;
  u.h[0] = *(const v8h*)(p);
  u.h[1] = *(const v8h*)(p + 16);
  return u.v;
}

// Same A layout, f32 source (attention weights) converted in registers.
__device__ __forceinline__ v16h load_frag_a(const float* __restrict__ base,
                                            int ld, int row0, int k0, int lane) {
  int m  = lane & 15;
  int hi = lane >> 4;
  const float* p = base + (size_t)(row0 + m) * ld + k0 + hi * 8;
  v4f f0 = *(const v4f*)(p);
  v4f f1 = *(const v4f*)(p + 4);
  v4f f2 = *(const v4f*)(p + 16);
  v4f f3 = *(const v4f*)(p + 20);
  v16h a;
#pragma unroll
  for (int i = 0; i < 4; ++i) {
    a[i]      = (half_t)f0[i];
    a[i + 4]  = (half_t)f1[i];
    a[i + 8]  = (half_t)f2[i];
    a[i + 12] = (half_t)f3[i];
  }
  return a;
}

// B fragment (32x16 KxN) from the LDS tile: lane l -> column n = l&15,
// k = (l>=16 ? 8 : 0) + 0..7, then +16 (two 16B ds loads).
__device__ __forceinline__ v16h load_frag_b_lds(const half_t* bbase, int n0, int lane) {
  int n  = lane & 15;
  int hi = lane >> 4;
  const half_t* p = bbase + (size_t)(n0 + n) * LDS_STRIDE + hi * 8;
  union { v16h v; v8h h[2]; } u;
  u.h[0] = *(const v8h*)(p);
  u.h[1] = *(const v8h*)(p + 16);
  return u.v;
}

// C/D (16x16 f32): lane l holds column n = l&15; VGPR r holds m = r + 8*(l>=16).
__device__ __forceinline__ void store_tile_f32(float* __restrict__ base, int ld,
                                               int row0, int col0, int lane,
                                               v8f c, float bias) {
  int n  = lane & 15;
  int mh = (lane >> 4) * 8;
  float* p = base + (size_t)(row0 + mh) * ld + col0 + n;
#pragma unroll
  for (int r = 0; r < 8; ++r) p[(size_t)r * ld] = c[r] + bias;
}

__device__ __forceinline__ void store_tile_f16(half_t* __restrict__ base, int ld,
                                               int row0, int col0, int lane,
                                               v8f c, float bias) {
  int n  = lane & 15;
  int mh = (lane >> 4) * 8;
  half_t* p = base + (size_t)(row0 + mh) * ld + col0 + n;
#pragma unroll
  for (int r = 0; r < 8; ++r) p[(size_t)r * ld] = (half_t)(c[r] + bias);
}

#define WMMA_F16(a, b, c) \
  __builtin_amdgcn_wmma_f32_16x16x32_f16(false, (a), false, (b), (short)0, (c), false, false)

// ---------------------------------------------------------------------------
// Core K-loop shared by all GEMMs.
//   A:   [*, lda] row-major (f16 or f32), this wave's rows start at rowA0 (32 rows)
//   Bsrc:[ldb-major] "S[n][k]" matrix (row n = output column n), k from bk0
//   Accumulates 8 C tiles: acc[nt] = rows rowA0..+15, acc[4+nt] = rows +16..+31
template <typename AT>
__device__ __forceinline__ void gemm_core(const AT* __restrict__ A, int lda, int rowA0,
                                          const half_t* __restrict__ Bsrc, int ldb,
                                          int bk0, int col0, int kTiles,
                                          half_t* Bs, v8f acc[8]) {
  const int tid    = threadIdx.x;
  const int lane   = tid & 31;
  const int lrow   = tid >> 2;                 // 0..63 : B-tile row this thread loads
  const int lchunk = tid & 3;                  // 0..3  : 8-half chunk within the row
  const half_t* gB = Bsrc + (size_t)(col0 + lrow) * ldb + bk0 + lchunk * 8;
  half_t* lB       = Bs + lrow * LDS_STRIDE + lchunk * 8;

  async_copy16(lB, gB);                        // preload k-tile 0 into buffer 0
  for (int kt = 0; kt < kTiles; ++kt) {
    const int cur = kt & 1;
    async_wait_all();
    __syncthreads();                           // LDS buffer 'cur' ready for all waves
    if (kt + 1 < kTiles)
      async_copy16(Bs + (cur ^ 1) * BUF_HALFS + lrow * LDS_STRIDE + lchunk * 8,
                   gB + (size_t)(kt + 1) * 32);
    v16h a0 = load_frag_a(A, lda, rowA0,      kt * 32, lane);
    v16h a1 = load_frag_a(A, lda, rowA0 + 16, kt * 32, lane);
    const half_t* bbase = Bs + cur * BUF_HALFS;
#pragma unroll
    for (int nt = 0; nt < 4; ++nt) {
      v16h bf = load_frag_b_lds(bbase, nt * 16, lane);
      acc[nt]     = WMMA_F16(a0, bf, acc[nt]);
      acc[4 + nt] = WMMA_F16(a1, bf, acc[4 + nt]);
    }
  }
  async_wait_all();
  __syncthreads();                             // LDS reusable by a following call
}

// ---------------------------------------------------------------------------
// Pack f32 -> f16 (grid-stride)
__global__ __launch_bounds__(256) void pack_f16_kernel(const float* __restrict__ src,
                                                       half_t* __restrict__ dst, size_t n) {
  size_t i = (size_t)blockIdx.x * blockDim.x + threadIdx.x;
  size_t stride = (size_t)gridDim.x * blockDim.x;
  for (; i < n; i += stride) dst[i] = (half_t)src[i];
}

// Transpose + pack V[b][l][d] (f32) -> VT[b][d][l] (f16), 32x32 LDS tiles
__global__ __launch_bounds__(256) void transpose_v_kernel(const float* __restrict__ V,
                                                          half_t* __restrict__ VT) {
  __shared__ half_t tile[32][33];
  int b  = blockIdx.z;
  int l0 = blockIdx.x * 32;
  int d0 = blockIdx.y * 32;
  const float* Vb = V + (size_t)b * LQ * DV;
  half_t*      Tb = VT + (size_t)b * DV * LQ;
  int tx = threadIdx.x, ty = threadIdx.y;      // block (32, 8)
#pragma unroll
  for (int j = 0; j < 4; ++j) {
    int l = ty + j * 8;
    tile[l][tx] = (half_t)Vb[(size_t)(l0 + l) * DV + d0 + tx];
  }
  __syncthreads();
#pragma unroll
  for (int j = 0; j < 4; ++j) {
    int d = ty + j * 8;
    Tb[(size_t)(d0 + d) * LQ + l0 + tx] = tile[tx][d];
  }
}

// ---------------------------------------------------------------------------
// GEMM 1: Qm[b] = Qh[b] (TQ x DQ) * Wqh^T + bq -> f16
__global__ __launch_bounds__(256) void gemm_qproj_kernel(const half_t* __restrict__ Qh,
                                                         const half_t* __restrict__ Wqh,
                                                         const float* __restrict__ bq,
                                                         half_t* __restrict__ Qmh) {
  __shared__ __align__(16) half_t Bs[2 * BUF_HALFS];
  int lane = threadIdx.x & 31, wave = threadIdx.x >> 5;
  int b    = blockIdx.z;
  int row0 = blockIdx.x * 256 + wave * 32;
  int col0 = blockIdx.y * 64;
  v8f acc[8] = {};
  gemm_core(Qh + (size_t)b * TQS * DQ, DQ, row0, Wqh, DQ, 0, col0, DQ / 32, Bs, acc);
  half_t* Out = Qmh + (size_t)b * TQS * DK;
#pragma unroll
  for (int nt = 0; nt < 4; ++nt) {
    float bias = bq[col0 + nt * 16 + (lane & 15)];
    store_tile_f16(Out, DK, row0,      col0 + nt * 16, lane, acc[nt],     bias);
    store_tile_f16(Out, DK, row0 + 16, col0 + nt * 16, lane, acc[4 + nt], bias);
  }
}

// GEMM 2: scores[b] = Qm[b] * Kh[b]^T -> f32 (directly into attn region of d_out)
__global__ __launch_bounds__(256) void gemm_scores_kernel(const half_t* __restrict__ Qmh,
                                                          const half_t* __restrict__ Kh,
                                                          float* __restrict__ attn) {
  __shared__ __align__(16) half_t Bs[2 * BUF_HALFS];
  int lane = threadIdx.x & 31, wave = threadIdx.x >> 5;
  int b    = blockIdx.z;
  int row0 = blockIdx.x * 256 + wave * 32;
  int col0 = blockIdx.y * 64;
  v8f acc[8] = {};
  gemm_core(Qmh + (size_t)b * TQS * DK, DK, row0,
            Kh + (size_t)b * LQ * DK, DK, 0, col0, DK / 32, Bs, acc);
  float* Out = attn + (size_t)b * TQS * LQ;
#pragma unroll
  for (int nt = 0; nt < 4; ++nt) {
    store_tile_f32(Out, LQ, row0,      col0 + nt * 16, lane, acc[nt],     0.0f);
    store_tile_f32(Out, LQ, row0 + 16, col0 + nt * 16, lane, acc[4 + nt], 0.0f);
  }
}

// Softmax in place over rows of length L=2048, one 256-thread block per row
__global__ __launch_bounds__(256) void softmax_kernel(float* __restrict__ attn) {
  float* p = attn + (size_t)blockIdx.x * LQ;
  int tid = threadIdx.x, lane = tid & 31, wave = tid >> 5;
  __shared__ float sred[8];
  __shared__ float sb;
  float v[8];
  float m = -3.4e38f;
#pragma unroll
  for (int i = 0; i < 8; ++i) { v[i] = p[tid + i * 256]; m = fmaxf(m, v[i]); }
#pragma unroll
  for (int off = 16; off; off >>= 1) m = fmaxf(m, __shfl_xor(m, off, 32));
  if (lane == 0) sred[wave] = m;
  __syncthreads();
  if (tid == 0) {
    float mm = sred[0];
#pragma unroll
    for (int i = 1; i < 8; ++i) mm = fmaxf(mm, sred[i]);
    sb = mm;
  }
  __syncthreads();
  m = sb;
  float s = 0.0f;
#pragma unroll
  for (int i = 0; i < 8; ++i) { v[i] = __expf(v[i] - m); s += v[i]; }
#pragma unroll
  for (int off = 16; off; off >>= 1) s += __shfl_xor(s, off, 32);
  __syncthreads();
  if (lane == 0) sred[wave] = s;
  __syncthreads();
  if (tid == 0) {
    float ss = 0.0f;
#pragma unroll
    for (int i = 0; i < 8; ++i) ss += sred[i];
    sb = 1.0f / ss;
  }
  __syncthreads();
  float inv = sb;
#pragma unroll
  for (int i = 0; i < 8; ++i) p[tid + i * 256] = v[i] * inv;
}

// GEMM 3: AA[b] = attn[b] (f32 -> f16 in regs) * V[b] -> f16 (B = VT rows)
__global__ __launch_bounds__(256) void gemm_av_kernel(const float* __restrict__ attn,
                                                      const half_t* __restrict__ VTh,
                                                      half_t* __restrict__ AAh) {
  __shared__ __align__(16) half_t Bs[2 * BUF_HALFS];
  int lane = threadIdx.x & 31, wave = threadIdx.x >> 5;
  int b    = blockIdx.z;
  int row0 = blockIdx.x * 256 + wave * 32;
  int col0 = blockIdx.y * 64;
  v8f acc[8] = {};
  gemm_core(attn + (size_t)b * TQS * LQ, LQ, row0,
            VTh + (size_t)b * DV * LQ, LQ, 0, col0, LQ / 32, Bs, acc);
  half_t* Out = AAh + (size_t)b * TQS * DV;
#pragma unroll
  for (int nt = 0; nt < 4; ++nt) {
    store_tile_f16(Out, DV, row0,      col0 + nt * 16, lane, acc[nt],     0.0f);
    store_tile_f16(Out, DV, row0 + 16, col0 + nt * 16, lane, acc[4 + nt], 0.0f);
  }
}

// GEMM 4: out[b] = [Qh[b], AA[b]] * Wch^T + bc -> f32 (concat folded: two K loops)
__global__ __launch_bounds__(256) void gemm_out_kernel(const half_t* __restrict__ Qh,
                                                       const half_t* __restrict__ AAh,
                                                       const half_t* __restrict__ Wch,
                                                       const float* __restrict__ bc,
                                                       float* __restrict__ out) {
  __shared__ __align__(16) half_t Bs[2 * BUF_HALFS];
  int lane = threadIdx.x & 31, wave = threadIdx.x >> 5;
  int b    = blockIdx.z;
  int row0 = blockIdx.x * 256 + wave * 32;
  int col0 = blockIdx.y * 64;
  const int LDW = DQ + DV;                     // 2048
  v8f acc[8] = {};
  gemm_core(Qh  + (size_t)b * TQS * DQ, DQ, row0, Wch, LDW, 0,  col0, DQ / 32, Bs, acc);
  gemm_core(AAh + (size_t)b * TQS * DV, DV, row0, Wch, LDW, DQ, col0, DV / 32, Bs, acc);
  float* Out = out + (size_t)b * TQS * DOUT;
#pragma unroll
  for (int nt = 0; nt < 4; ++nt) {
    float bias = bc[col0 + nt * 16 + (lane & 15)];
    store_tile_f32(Out, DOUT, row0,      col0 + nt * 16, lane, acc[nt],     bias);
    store_tile_f32(Out, DOUT, row0 + 16, col0 + nt * 16, lane, acc[4 + nt], bias);
  }
}

// ---------------------------------------------------------------------------
extern "C" void kernel_launch(void* const* d_in, const int* in_sizes, int n_in,
                              void* d_out, int out_size, void* d_ws, size_t ws_size,
                              hipStream_t stream) {
  const float* Q  = (const float*)d_in[0];  // [B,TQ,DQ]
  const float* K  = (const float*)d_in[1];  // [B,L,DK]
  const float* V  = (const float*)d_in[2];  // [B,L,DV]
  const float* Wq = (const float*)d_in[3];  // [DK,DQ]
  const float* bq = (const float*)d_in[4];  // [DK]
  const float* Wc = (const float*)d_in[5];  // [DOUT,DQ+DV]
  const float* bc = (const float*)d_in[6];  // [DOUT]

  float* out  = (float*)d_out;                                  // [B,TQ,DOUT]
  float* attn = (float*)d_out + (size_t)BQ * TQS * DOUT;        // [B,TQ,L]

  // Workspace layout (f16 staging), ~454 MB total
  char* ws = (char*)d_ws;
  const size_t nQ  = (size_t)BQ * TQS * DQ;
  const size_t nK  = (size_t)BQ * LQ * DK;
  const size_t nV  = (size_t)BQ * LQ * DV;
  const size_t nWq = (size_t)DK * DQ;
  const size_t nWc = (size_t)DOUT * (DQ + DV);
  const size_t nQm = (size_t)BQ * TQS * DK;
  const size_t nAA = (size_t)BQ * TQS * DV;

  half_t* Qh  = (half_t*)ws;                 ws += nQ  * sizeof(half_t);
  half_t* Kh  = (half_t*)ws;                 ws += nK  * sizeof(half_t);
  half_t* VTh = (half_t*)ws;                 ws += nV  * sizeof(half_t);
  half_t* Wqh = (half_t*)ws;                 ws += nWq * sizeof(half_t);
  half_t* Wch = (half_t*)ws;                 ws += nWc * sizeof(half_t);
  half_t* Qmh = (half_t*)ws;                 ws += nQm * sizeof(half_t);
  half_t* AAh = (half_t*)ws;                 ws += nAA * sizeof(half_t);

  // 1) pack to f16
  pack_f16_kernel<<<8192, 256, 0, stream>>>(Q,  Qh,  nQ);
  pack_f16_kernel<<<8192, 256, 0, stream>>>(K,  Kh,  nK);
  pack_f16_kernel<<<1024, 256, 0, stream>>>(Wq, Wqh, nWq);
  pack_f16_kernel<<<2048, 256, 0, stream>>>(Wc, Wch, nWc);
  transpose_v_kernel<<<dim3(LQ / 32, DV / 32, BQ), dim3(32, 8), 0, stream>>>(V, VTh);

  // 2) Q projection
  gemm_qproj_kernel<<<dim3(TQS / 256, DK / 64, BQ), 256, 0, stream>>>(Qh, Wqh, bq, Qmh);

  // 3) scores -> attn region of d_out
  gemm_scores_kernel<<<dim3(TQS / 256, LQ / 64, BQ), 256, 0, stream>>>(Qmh, Kh, attn);

  // 4) softmax in place (these are the returned attn_weights)
  softmax_kernel<<<BQ * TQS, 256, 0, stream>>>(attn);

  // 5) attn @ V
  gemm_av_kernel<<<dim3(TQS / 256, DV / 64, BQ), 256, 0, stream>>>(attn, VTh, AAh);

  // 6) output projection over the (folded) concat
  gemm_out_kernel<<<dim3(TQS / 256, DOUT / 64, BQ), 256, 0, stream>>>(Qh, AAh, Wch, bc, out);
}
